// MultiheadAttention_35734127903448
// MI455X (gfx1250) — compile-verified, matrix operand order checked by reading, not run on hardware
//
#include <hip/hip_runtime.h>
#include <hip/hip_bf16.h>

// ---- problem constants (from reference) ----
#define B_    4
#define S_    2048
#define E_    1024
#define H_    16
#define DK_   64
#define ROWS_ (B_ * S_)     // 8192
#define HD_   (H_ * DK_)    // 1024

typedef __bf16 bf16;
typedef __attribute__((ext_vector_type(16))) __bf16 bf16x16;
typedef __attribute__((ext_vector_type(8)))  __bf16 bf16x8;
typedef __attribute__((ext_vector_type(8)))  float  f32x8;
typedef __attribute__((ext_vector_type(4)))  float  f32x4;

// ---- Tensor Data Mover availability (compile-safe guards) ----
#if defined(__has_builtin)
#  if __has_builtin(__builtin_amdgcn_tensor_load_to_lds) && \
      __has_builtin(__builtin_amdgcn_s_wait_tensorcnt)
#    define HAVE_TDM 1
#  endif
#endif
#ifndef HAVE_TDM
#  define HAVE_TDM 0
#endif
// ROCm 7.2 clang-22: 5-arg builtin; amdgpu-toolchain (ships TDM header): 6-arg.
#if defined(__has_include)
#  if __has_include(<hip/amd_detail/amd_gfx1250_TDM.h>)
#    define TDM_ARGS6 1
#  endif
#endif
#ifndef TDM_ARGS6
#  define TDM_ARGS6 0
#endif

typedef unsigned int u32x4 __attribute__((ext_vector_type(4)));
typedef int          i32x4 __attribute__((ext_vector_type(4)));
typedef int          i32x8 __attribute__((ext_vector_type(8)));

static __device__ __forceinline__ f32x8 zero8() {
  f32x8 z = {0.f, 0.f, 0.f, 0.f, 0.f, 0.f, 0.f, 0.f};
  return z;
}

static __device__ __forceinline__ f32x8 wmma_bf16(bf16x16 a, bf16x16 b, f32x8 c) {
  // D = A(16x32 bf16) x B(32x16 bf16) + C(16x16 f32)
  return __builtin_amdgcn_wmma_f32_16x16x32_bf16(
      /*neg_a=*/false, a, /*neg_b=*/false, b,
      /*c_mod=*/(short)0, c, /*reuse_a=*/false, /*reuse_b=*/false);
}

union FragU { bf16x16 v; bf16x8 h[2]; };

// Load a 16x32 A-fragment (or transposed-B fragment) from an LDS row.
// Per-lane layout (ISA 7.12.2, 16-bit A 16x32): lane<16 holds K 0..7 & 16..23,
// lane>=16 holds K 8..15 & 24..31  =>  chunks at (kb) and (kb+16), kb=(lane>>4)*8.
static __device__ __forceinline__ bf16x16 ld_frag(const bf16* row, int kb) {
  FragU u;
  u.h[0] = *(const bf16x8*)(row + kb);
  u.h[1] = *(const bf16x8*)(row + kb + 16);
  return u.v;
}

// ---------------- GEMM tiling ----------------
#define BM  128
#define BN  128
#define BKT 32
#define LDA 40   // padded LDS row length in bf16 (80B stride, 16B aligned)

// ============ Kernel 1: fused QKV projection ============
// X(8192x1024) @ W(1024x1024) + b, one of {Q,K,V} per blockIdx.z.
// Output bf16 in (B,H,S,DK) layout.
__global__ __launch_bounds__(256) void qkv_kernel(
    const float* __restrict__ X,
    const float* __restrict__ Wq, const float* __restrict__ bq,
    const float* __restrict__ Wk, const float* __restrict__ bk,
    const float* __restrict__ Wv, const float* __restrict__ bv,
    bf16* __restrict__ qw, bf16* __restrict__ kw, bf16* __restrict__ vw)
{
  const float* W; const float* bias; bf16* out;
  if (blockIdx.z == 0)      { W = Wq; bias = bq; out = qw; }
  else if (blockIdx.z == 1) { W = Wk; bias = bk; out = kw; }
  else                      { W = Wv; bias = bv; out = vw; }

  __shared__ bf16 As[BM * LDA];   // [row][k]
  __shared__ bf16 Bs[BN * LDA];   // [col][k]  (W tile staged transposed)

  const int tid  = threadIdx.x;
  const int lane = tid & 31;
  const int wave = tid >> 5;      // 0..7
  const int wm   = wave & 3;      // 4 waves along M (32 rows each)
  const int wn   = wave >> 2;     // 2 waves along N (64 cols each)
  const int m0   = blockIdx.x * BM;
  const int n0   = blockIdx.y * BN;
  const int kb   = (lane >> 4) * 8;
  const int ln   = lane & 15;

  f32x8 acc[2][4];
#pragma unroll
  for (int i = 0; i < 2; i++)
#pragma unroll
    for (int j = 0; j < 4; j++) acc[i][j] = zero8();

  for (int k0 = 0; k0 < E_; k0 += BKT) {
    __syncthreads();
    // stage A: 128x32 f32 -> bf16
#pragma unroll
    for (int i = 0; i < 4; i++) {
      int e   = tid + i * 256;        // 0..1023 float4 chunks
      int row = e >> 3;               // 8 float4 per row
      int c4  = (e & 7) * 4;
      f32x4 v = *(const f32x4*)(X + (size_t)(m0 + row) * E_ + k0 + c4);
      bf16* d = As + row * LDA + c4;
      d[0] = (bf16)v[0]; d[1] = (bf16)v[1]; d[2] = (bf16)v[2]; d[3] = (bf16)v[3];
    }
    // stage B transposed: W[k0+kr][n0+nc] -> Bs[nc][kr]
#pragma unroll
    for (int i = 0; i < 4; i++) {
      int e  = tid + i * 256;
      int kr = e >> 5;                // 32 float4 per row of 128
      int c4 = (e & 31) * 4;
      f32x4 v = *(const f32x4*)(W + (size_t)(k0 + kr) * HD_ + n0 + c4);
      Bs[(c4 + 0) * LDA + kr] = (bf16)v[0];
      Bs[(c4 + 1) * LDA + kr] = (bf16)v[1];
      Bs[(c4 + 2) * LDA + kr] = (bf16)v[2];
      Bs[(c4 + 3) * LDA + kr] = (bf16)v[3];
    }
    __syncthreads();

    bf16x16 afr[2], bfr[4];
#pragma unroll
    for (int ms = 0; ms < 2; ms++)
      afr[ms] = ld_frag(As + (wm * 32 + ms * 16 + ln) * LDA, kb);
#pragma unroll
    for (int ns = 0; ns < 4; ns++)
      bfr[ns] = ld_frag(Bs + (wn * 64 + ns * 16 + ln) * LDA, kb);
#pragma unroll
    for (int ms = 0; ms < 2; ms++)
#pragma unroll
      for (int ns = 0; ns < 4; ns++)
        acc[ms][ns] = wmma_bf16(afr[ms], bfr[ns], acc[ms][ns]);
  }

  // epilogue: +bias, store bf16 into (B,H,S,DK)
  const int mh = (lane >> 4) * 8;   // C layout: lane<16 -> M=r ; lane>=16 -> M=r+8
#pragma unroll
  for (int ms = 0; ms < 2; ms++) {
#pragma unroll
    for (int ns = 0; ns < 4; ns++) {
      int col = n0 + wn * 64 + ns * 16 + ln;
      int h = col >> 6, d = col & 63;
      float bvv = bias[col];
#pragma unroll
      for (int r = 0; r < 8; r++) {
        int row = m0 + wm * 32 + ms * 16 + r + mh;
        int b = row >> 11, s = row & (S_ - 1);
        out[(((size_t)(b * H_ + h)) * S_ + s) * DK_ + d] =
            (bf16)(acc[ms][ns][r] + bvv);
      }
    }
  }
}

// ============ Kernel 2: flash attention ============
#define QT  128   // query rows per block (8 waves x 16 rows)
#define KT  64    // keys per tile
#define LDKp 72   // padded LDS row (144B stride, 16B aligned)

__global__ __launch_bounds__(256) void attn_kernel(
    const bf16* __restrict__ qw, const bf16* __restrict__ kw,
    const bf16* __restrict__ vw, bf16* __restrict__ ctx)
{
  // NOTE: Ks is deliberately UNPADDED (row stride 64*2=128B) so the TDM can
  // write it as a dense 64x64 tile; 128B stride is still 16B aligned for
  // b128 fragment reads.
  __shared__ bf16 Ks[KT * DK_];         // [key][d]  (B-frag for Q@K^T = row of K)
  __shared__ bf16 Vt[DK_ * LDKp];       // [d][key]  (B-frag for P@V = column of V)
  __shared__ bf16 Pb[8 * 16 * LDKp];    // per-wave P tile, 16x64

  const int tid  = threadIdx.x;
  const int lane = tid & 31;
  const int wave = tid >> 5;
  const int bh   = blockIdx.y;                 // b*H + h
  const int q0   = blockIdx.x * QT + wave * 16;
  const size_t base = (size_t)bh * S_ * DK_;
  const int kb = (lane >> 4) * 8;
  const int ln = lane & 15;
  const int mh = (lane >> 4) * 8;

  // Q fragments for both 32-wide k-steps of DK=64 (direct from global, b128s)
  bf16x16 qf[2];
#pragma unroll
  for (int ks = 0; ks < 2; ks++) {
    const bf16* row = qw + base + (size_t)(q0 + ln) * DK_ + ks * 32;
    FragU u;
    u.h[0] = *(const bf16x8*)(row + kb);
    u.h[1] = *(const bf16x8*)(row + kb + 16);
    qf[ks] = u.v;
  }

  f32x8 o[4];
#pragma unroll
  for (int d = 0; d < 4; d++) o[d] = zero8();
  float mrow[8], lrow[8];
#pragma unroll
  for (int r = 0; r < 8; r++) { mrow[r] = -3.0e38f; lrow[r] = 0.f; }

  const float scale = 0.125f;       // 1/sqrt(64)
  const float LOG2E = 1.44269504f;
  bf16* myP = Pb + wave * 16 * LDKp;

#if HAVE_TDM
  // Invariant parts of the Tensor DMA descriptor (ISA cdna5 §8.3/8.4).
  // 2-D tensor: K for this (b,h) = 2048 x 64 bf16, tile = 64 x 64.
  const unsigned lds_ks = (unsigned)(size_t)(&Ks[0]);   // LDS byte offset
  i32x8 g1;
  g1[0] = 0x00010000;          // workgroup_mask=0, data_size=1 (2 bytes)
  g1[1] = (int)(64u << 16);    // tensor_dim0[15:0]=64 (bits 63:48)
  g1[2] = (int)(2048u << 16);  // tensor_dim0[31:16]=0, tensor_dim1[15:0]=2048
  g1[3] = (int)(64u << 16);    // tensor_dim1[31:16]=0, tile_dim0=64
  g1[4] = 64;                  // tile_dim1=64, tile_dim2=0
  g1[5] = 64;                  // tensor_dim0_stride[31:0]=64
  g1[6] = 0;                   // tensor_dim0_stride[47:32]=0, dim1_stride lo=0
  g1[7] = 0;                   // tensor_dim1_stride hi = 0
  i32x4 gz; gz[0] = 0; gz[1] = 0; gz[2] = 0; gz[3] = 0;
#if TDM_ARGS6
  i32x8 gz8;
#pragma unroll
  for (int i = 0; i < 8; i++) gz8[i] = 0;
#endif
#endif

  for (int kt = 0; kt < S_; kt += KT) {
    __syncthreads();

#if HAVE_TDM
    // --- K tile via Tensor Data Mover: dense 64x64 bf16 global -> LDS ---
    if (wave == 0) {
      unsigned long long ga =
          (unsigned long long)(size_t)(kw + base + (size_t)kt * DK_);
      u32x4 g0;
      g0[0] = 1u;                                   // count=1 (valid user D#)
      g0[1] = lds_ks;                               // lds_addr (bytes)
      g0[2] = (unsigned)(ga & 0xFFFFFFFFu);         // global_addr[31:0]
      g0[3] = (unsigned)((ga >> 32) & 0x01FFFFFFu)  // global_addr[56:32]
              | (2u << 30);                         // type=2 ("image")
#if TDM_ARGS6
      __builtin_amdgcn_tensor_load_to_lds(g0, g1, gz, gz, gz8, 0);
#else
      __builtin_amdgcn_tensor_load_to_lds(g0, g1, gz, gz, 0);
#endif
    }
#endif

    // --- V^T tile [d][key] staged manually (TDM cannot transpose) ---
#pragma unroll
    for (int i = 0; i < 2; i++) {
      int c    = tid * 2 + i;         // 0..511 chunks of 8 bf16
      int krow = c >> 3;              // key 0..63
      int coff = (c & 7) * 8;         // d offset
      bf16x8 vv = *(const bf16x8*)(vw + base + (size_t)(kt + krow) * DK_ + coff);
#pragma unroll
      for (int j = 0; j < 8; j++) Vt[(coff + j) * LDKp + krow] = vv[j];
#if !HAVE_TDM
      bf16x8 kv = *(const bf16x8*)(kw + base + (size_t)(kt + krow) * DK_ + coff);
      *(bf16x8*)(Ks + krow * DK_ + coff) = kv;
#endif
    }
    // prefetch next K/V tile into cache (global_prefetch_b8)
    if (kt + KT < S_) {
      __builtin_prefetch(kw + base + (size_t)(kt + KT) * DK_ + tid * 16, 0, 0);
      __builtin_prefetch(vw + base + (size_t)(kt + KT) * DK_ + tid * 16, 0, 0);
    }
#if HAVE_TDM
    if (wave == 0) __builtin_amdgcn_s_wait_tensorcnt(0);
#endif
    __syncthreads();

    // ---- S = (Q @ K^T) * scale : 4 key-subtiles of 16 ----
    f32x8 sacc[4];
#pragma unroll
    for (int n = 0; n < 4; n++) sacc[n] = zero8();
#pragma unroll
    for (int n = 0; n < 4; n++)
#pragma unroll
      for (int ks = 0; ks < 2; ks++) {
        bf16x16 bfrag = ld_frag(Ks + (n * 16 + ln) * DK_, kb + ks * 32);
        sacc[n] = wmma_bf16(qf[ks], bfrag, sacc[n]);
      }
#pragma unroll
    for (int n = 0; n < 4; n++)
#pragma unroll
      for (int r = 0; r < 8; r++) sacc[n][r] *= scale;

    // ---- online softmax (rows live in 16-lane halves; reduce via shfl_xor) ----
    float alpha[8];
#pragma unroll
    for (int r = 0; r < 8; r++) {
      float mx = fmaxf(fmaxf(sacc[0][r], sacc[1][r]),
                       fmaxf(sacc[2][r], sacc[3][r]));
#pragma unroll
      for (int off = 1; off < 16; off <<= 1)
        mx = fmaxf(mx, __shfl_xor(mx, off, 32));
      float mnew = fmaxf(mrow[r], mx);
      alpha[r] = exp2f((mrow[r] - mnew) * LOG2E);
      mrow[r] = mnew;
    }
#pragma unroll
    for (int n = 0; n < 4; n++)
#pragma unroll
      for (int r = 0; r < 8; r++)
        sacc[n][r] = exp2f((sacc[n][r] - mrow[r]) * LOG2E);
#pragma unroll
    for (int r = 0; r < 8; r++) {
      float ls = sacc[0][r] + sacc[1][r] + sacc[2][r] + sacc[3][r];
#pragma unroll
      for (int off = 1; off < 16; off <<= 1)
        ls += __shfl_xor(ls, off, 32);
      lrow[r] = lrow[r] * alpha[r] + ls;
    }
#pragma unroll
    for (int d = 0; d < 4; d++)
#pragma unroll
      for (int r = 0; r < 8; r++) o[d][r] *= alpha[r];

    // ---- P (bf16) to per-wave LDS, then O += P @ V ----
#pragma unroll
    for (int n = 0; n < 4; n++)
#pragma unroll
      for (int r = 0; r < 8; r++)
        myP[(r + mh) * LDKp + n * 16 + ln] = (bf16)sacc[n][r];
    // same-wave LDS ops are in-order; no barrier needed before re-reading myP

    bf16x16 paf[2];
#pragma unroll
    for (int ks = 0; ks < 2; ks++)
      paf[ks] = ld_frag(myP + ln * LDKp, kb + ks * 32);
#pragma unroll
    for (int d = 0; d < 4; d++)
#pragma unroll
      for (int ks = 0; ks < 2; ks++) {
        bf16x16 bfrag = ld_frag(Vt + (d * 16 + ln) * LDKp, kb + ks * 32);
        o[d] = wmma_bf16(paf[ks], bfrag, o[d]);
      }
  }

  // epilogue: normalize, store ctx bf16 in (B,S,H,DK)
  const int b = bh >> 4, h = bh & 15;
#pragma unroll
  for (int d4 = 0; d4 < 4; d4++) {
    int d = d4 * 16 + ln;
#pragma unroll
    for (int r = 0; r < 8; r++) {
      int s = q0 + r + mh;
      float inv = 1.f / lrow[r];
      ctx[(((size_t)(b * S_ + s)) * H_ + h) * DK_ + d] = (bf16)(o[d4][r] * inv);
    }
  }
}

// ============ Kernel 3: output projection ============
// ctx(8192x1024 bf16) @ Wo(1024x1024 f32) + bo -> out f32
__global__ __launch_bounds__(256) void oproj_kernel(
    const bf16* __restrict__ A, const float* __restrict__ Wo,
    const float* __restrict__ bo, float* __restrict__ out)
{
  __shared__ bf16 As[BM * LDA];
  __shared__ bf16 Bs[BN * LDA];

  const int tid  = threadIdx.x;
  const int lane = tid & 31;
  const int wave = tid >> 5;
  const int wm   = wave & 3;
  const int wn   = wave >> 2;
  const int m0   = blockIdx.x * BM;
  const int n0   = blockIdx.y * BN;
  const int kb   = (lane >> 4) * 8;
  const int ln   = lane & 15;

  f32x8 acc[2][4];
#pragma unroll
  for (int i = 0; i < 2; i++)
#pragma unroll
    for (int j = 0; j < 4; j++) acc[i][j] = zero8();

  for (int k0 = 0; k0 < HD_; k0 += BKT) {
    __syncthreads();
    // stage A (already bf16): 128x32, b128 copies
#pragma unroll
    for (int i = 0; i < 2; i++) {
      int c   = tid * 2 + i;         // 0..511
      int row = c >> 2;              // 4 chunks of 8 per 32-wide row
      int off = (c & 3) * 8;
      *(bf16x8*)(As + row * LDA + off) =
          *(const bf16x8*)(A + (size_t)(m0 + row) * HD_ + k0 + off);
    }
    // stage Wo transposed f32->bf16
#pragma unroll
    for (int i = 0; i < 4; i++) {
      int e  = tid + i * 256;
      int kr = e >> 5;
      int c4 = (e & 31) * 4;
      f32x4 v = *(const f32x4*)(Wo + (size_t)(k0 + kr) * E_ + n0 + c4);
      Bs[(c4 + 0) * LDA + kr] = (bf16)v[0];
      Bs[(c4 + 1) * LDA + kr] = (bf16)v[1];
      Bs[(c4 + 2) * LDA + kr] = (bf16)v[2];
      Bs[(c4 + 3) * LDA + kr] = (bf16)v[3];
    }
    __syncthreads();

    bf16x16 afr[2], bfr[4];
#pragma unroll
    for (int ms = 0; ms < 2; ms++)
      afr[ms] = ld_frag(As + (wm * 32 + ms * 16 + ln) * LDA, kb);
#pragma unroll
    for (int ns = 0; ns < 4; ns++)
      bfr[ns] = ld_frag(Bs + (wn * 64 + ns * 16 + ln) * LDA, kb);
#pragma unroll
    for (int ms = 0; ms < 2; ms++)
#pragma unroll
      for (int ns = 0; ns < 4; ns++)
        acc[ms][ns] = wmma_bf16(afr[ms], bfr[ns], acc[ms][ns]);
  }

  const int mh = (lane >> 4) * 8;
#pragma unroll
  for (int ms = 0; ms < 2; ms++) {
#pragma unroll
    for (int ns = 0; ns < 4; ns++) {
      int col = n0 + wn * 64 + ns * 16 + ln;
      float bvv = bo[col];
#pragma unroll
      for (int r = 0; r < 8; r++) {
        int row = m0 + wm * 32 + ms * 16 + r + mh;
        out[(size_t)row * E_ + col] = acc[ms][ns][r] + bvv;
      }
    }
  }
}

// ============ launcher ============
extern "C" void kernel_launch(void* const* d_in, const int* in_sizes, int n_in,
                              void* d_out, int out_size, void* d_ws, size_t ws_size,
                              hipStream_t stream) {
  (void)in_sizes; (void)n_in; (void)out_size; (void)ws_size;
  const float* X  = (const float*)d_in[0];
  const float* Wq = (const float*)d_in[1];
  const float* bq = (const float*)d_in[2];
  const float* Wk = (const float*)d_in[3];
  const float* bk = (const float*)d_in[4];
  const float* Wv = (const float*)d_in[5];
  const float* bv = (const float*)d_in[6];
  const float* Wo = (const float*)d_in[7];
  const float* bo = (const float*)d_in[8];
  float* out = (float*)d_out;

  const size_t mat = (size_t)ROWS_ * HD_ * sizeof(bf16);  // 16 MB each
  char* ws = (char*)d_ws;
  bf16* qw = (bf16*)(ws + 0 * mat);
  bf16* kw = (bf16*)(ws + 1 * mat);
  bf16* vw = (bf16*)(ws + 2 * mat);
  bf16* cw = (bf16*)(ws + 3 * mat);

  qkv_kernel<<<dim3(ROWS_ / BM, HD_ / BN, 3), 256, 0, stream>>>(
      X, Wq, bq, Wk, bk, Wv, bv, qw, kw, vw);
  attn_kernel<<<dim3(S_ / QT, B_ * H_), 256, 0, stream>>>(qw, kw, vw, cw);
  oproj_kernel<<<dim3(ROWS_ / BM, E_ / BN), 256, 0, stream>>>(cw, Wo, bo, out);
}
